// ReconstructEnvir_13280038879914
// MI455X (gfx1250) — matrix operand order repaired and checked by reading, not compiled, not run on hardware
//
#include <hip/hip_runtime.h>

#define Bv 1024
#define Tv 50
#define Fv 256
#define Hv 256
#define PV 10
#define BH (Bv * Hv) // elements in one (B,H) slab

typedef __bf16 bf16_t;
typedef __attribute__((ext_vector_type(16))) __bf16 v16bf;
typedef __attribute__((ext_vector_type(8)))  __bf16 v8bf;
typedef __attribute__((ext_vector_type(8)))  float  v8f;

__device__ __forceinline__ bf16_t f2bf(float f) {
  union { float f; unsigned u; } v; v.f = f;
  unsigned r = v.u + 0x7FFFu + ((v.u >> 16) & 1u); // round-to-nearest-even
  union { unsigned short s; bf16_t b; } o; o.s = (unsigned short)(r >> 16);
  return o.b;
}

// Fast gate math: v_exp_f32 + v_rcp_f32 instead of libm tanh / IEEE divide.
__device__ __forceinline__ float fast_sigm(float x) {
  return __builtin_amdgcn_rcpf(1.0f + __expf(-x));
}
__device__ __forceinline__ float fast_tanh(float x) {
  return 2.0f * __builtin_amdgcn_rcpf(1.0f + __expf(-2.0f * x)) - 1.0f;
}

// A fragment (16x32 bf16): per lane 16 contiguous K of row (tileM+lm):
//   K = k0 + 16*half .. +15
// A operands are single-use streams per step -> non-temporal (keep weights hot in L2).
__device__ __forceinline__ v16bf load_a_frag(const bf16_t* rowBase, int k0, int half) {
  union { v8bf h[2]; v16bf v; } u;
  u.h[0] = __builtin_nontemporal_load((const v8bf*)(rowBase + k0 + 16 * half));
  u.h[1] = __builtin_nontemporal_load((const v8bf*)(rowBase + k0 + 16 * half + 8));
  return u.v;
}
// B fragment (32x16 bf16): per lane 16 contiguous K of column (tileN+lm),
// lanes 0-15 -> K=k0..k0+15, lanes 16-31 -> K=k0+16..k0+31 (weights stored N-major).
// Weights are re-read every step: default (RT) temporal policy.
__device__ __forceinline__ v16bf load_b_frag(const bf16_t* rowBase, int k0, int half) {
  return *(const v16bf*)(rowBase + k0 + 16 * half);
}

// --- one 256-wide K segment: 8 k-chunks x 4 gate WMMAs, A fragment shared ---
__device__ __forceinline__ void gemm_seg(const bf16_t* __restrict__ aRow,
                                         const bf16_t* __restrict__ w0,
                                         const bf16_t* __restrict__ w1,
                                         const bf16_t* __restrict__ w2,
                                         const bf16_t* __restrict__ w3,
                                         int kb, int half,
                                         v8f& ai, v8f& af, v8f& ag, v8f& ao) {
#pragma unroll
  for (int k0 = 0; k0 < 256; k0 += 32) {
    v16bf a  = load_a_frag(aRow, k0, half);
    v16bf b0 = load_b_frag(w0, kb + k0, half);
    v16bf b1 = load_b_frag(w1, kb + k0, half);
    v16bf b2 = load_b_frag(w2, kb + k0, half);
    v16bf b3 = load_b_frag(w3, kb + k0, half);
    ai = __builtin_amdgcn_wmma_f32_16x16x32_bf16(false, a, false, b0, (short)0, ai, false, false);
    af = __builtin_amdgcn_wmma_f32_16x16x32_bf16(false, a, false, b1, (short)0, af, false, false);
    ag = __builtin_amdgcn_wmma_f32_16x16x32_bf16(false, a, false, b2, (short)0, ag, false, false);
    ao = __builtin_amdgcn_wmma_f32_16x16x32_bf16(false, a, false, b3, (short)0, ao, false, false);
  }
}

// ---------------- one-time conversion kernels ----------------

// (B,T,F) f32 -> (T,B,F) bf16  (time-major so each step slice is contiguous)
__global__ void convert_X_kernel(const float* __restrict__ src, bf16_t* __restrict__ dst) {
  int idx = blockIdx.x * blockDim.x + threadIdx.x;
  if (idx >= Bv * Tv * Fv) return;
  int b = idx / (Tv * Fv);
  int r = idx % (Tv * Fv);
  int t = r / Fv;
  int f = r % Fv;
  float x = __builtin_nontemporal_load(src + idx);
  dst[(size_t)t * (Bv * Fv) + (size_t)b * Fv + f] = f2bf(x);
}

// src: Ksrc x N (f32, row-major) -> dst[n*dstStride + kOff + k] = src[k][n] (bf16)
__global__ void transpose_cvt_kernel(const float* __restrict__ src, bf16_t* __restrict__ dst,
                                     int Ksrc, int N, int dstStride, int kOff) {
  int idx = blockIdx.x * blockDim.x + threadIdx.x;
  if (idx >= Ksrc * N) return;
  int k = idx / N, n = idx % N;
  float x = __builtin_nontemporal_load(src + idx);
  dst[(size_t)n * dstStride + kOff + k] = f2bf(x);
}

// ---------------- fused LSTM step ----------------
// z = [A0|A1|A2] @ Wt^T + b ; gates; c,h update.  Wt is 1024 rows (4 gates x 256 N) x K bf16.
// grid(8,16) x 256 threads: block tile 64(M) x 32(Ngate); wave tile 16x16 per gate.
__global__ __launch_bounds__(256) void lstm_step_kernel(
    const bf16_t* __restrict__ A0, const bf16_t* __restrict__ A1, const bf16_t* __restrict__ A2,
    const bf16_t* __restrict__ Wt, int K,
    const float* __restrict__ bias,
    float* __restrict__ cbuf,
    bf16_t* __restrict__ hOut)
{
  const int wave = threadIdx.x >> 5;
  const int lane = threadIdx.x & 31;
  const int half = lane >> 4;
  const int lm   = lane & 15;
  const int tileM = blockIdx.y * 64 + (wave & 3) * 16;
  const int tileN = blockIdx.x * 32 + (wave >> 2) * 16;

  v8f acc_i = {}, acc_f = {}, acc_g = {}, acc_o = {};

  const size_t rowA = (size_t)(tileM + lm) * 256;
  const bf16_t* w0 = Wt + (size_t)(tileN + lm) * K;         // gate i
  const bf16_t* w1 = Wt + (size_t)(256 + tileN + lm) * K;   // gate f
  const bf16_t* w2 = Wt + (size_t)(512 + tileN + lm) * K;   // gate g
  const bf16_t* w3 = Wt + (size_t)(768 + tileN + lm) * K;   // gate o

  gemm_seg(A0 + rowA, w0, w1, w2, w3, 0,   half, acc_i, acc_f, acc_g, acc_o);
  gemm_seg(A1 + rowA, w0, w1, w2, w3, 256, half, acc_i, acc_f, acc_g, acc_o);
  if (A2)
    gemm_seg(A2 + rowA, w0, w1, w2, w3, 512, half, acc_i, acc_f, acc_g, acc_o);

  const int n = tileN + lm;
  const float bi = bias[n], bff = bias[256 + n], bg = bias[512 + n], bo = bias[768 + n];
#pragma unroll
  for (int r = 0; r < 8; ++r) {
    const int m = tileM + r + half * 8;
    const size_t idx = (size_t)m * Hv + n;
    const float zi = acc_i[r] + bi;
    const float zf = acc_f[r] + bff;
    const float zg = acc_g[r] + bg;
    const float zo = acc_o[r] + bo;
    const float cold = cbuf[idx];
    const float cn = fast_sigm(zf) * cold + fast_sigm(zi) * fast_tanh(zg);
    const float hn = fast_sigm(zo) * fast_tanh(cn);
    cbuf[idx] = cn;
    hOut[idx] = f2bf(hn);
  }
}

// ---------------- dense + ReLU (M=1024, N=256, K=256) ----------------
__global__ __launch_bounds__(256) void dense_relu_kernel(
    const bf16_t* __restrict__ A, const bf16_t* __restrict__ Wt,
    const float* __restrict__ bias,
    bf16_t* __restrict__ outBf, float* __restrict__ outF)
{
  const int wave = threadIdx.x >> 5;
  const int lane = threadIdx.x & 31;
  const int half = lane >> 4;
  const int lm   = lane & 15;
  const int tileM = blockIdx.y * 64 + (wave & 3) * 16;
  const int tileN = blockIdx.x * 32 + (wave >> 2) * 16;

  v8f acc = {};
  const bf16_t* aRow = A  + (size_t)(tileM + lm) * 256;
  const bf16_t* wRow = Wt + (size_t)(tileN + lm) * 256;
#pragma unroll
  for (int k0 = 0; k0 < 256; k0 += 32) {
    v16bf a = load_a_frag(aRow, k0, half);
    v16bf b = load_b_frag(wRow, k0, half);
    acc = __builtin_amdgcn_wmma_f32_16x16x32_bf16(false, a, false, b, (short)0, acc, false, false);
  }
  const int n = tileN + lm;
  const float bb = bias[n];
#pragma unroll
  for (int r = 0; r < 8; ++r) {
    const int m = tileM + r + half * 8;
    const size_t idx = (size_t)m * 256 + n;
    float v = acc[r] + bb;
    v = fmaxf(v, 0.0f);
    if (outF) outF[idx] = v;
    else      outBf[idx] = f2bf(v);
  }
}

// ---------------- host ----------------
extern "C" void kernel_launch(void* const* d_in, const int* in_sizes, int n_in,
                              void* d_out, int out_size, void* d_ws, size_t ws_size,
                              hipStream_t stream) {
  (void)in_sizes; (void)n_in; (void)out_size; (void)ws_size;
  const float* input = (const float*)d_in[0];
  const float* enc_W = (const float*)d_in[1];
  const float* enc_U = (const float*)d_in[2];
  const float* enc_b = (const float*)d_in[3];
  const float* dec_W = (const float*)d_in[4];
  const float* dec_U = (const float*)d_in[5];
  const float* dec_b = (const float*)d_in[6];
  const float* d1_W  = (const float*)d_in[7];
  const float* d1_b  = (const float*)d_in[8];
  const float* d2_W  = (const float*)d_in[9];
  const float* d2_b  = (const float*)d_in[10];
  const float* d3_W  = (const float*)d_in[11];
  const float* d3_b  = (const float*)d_in[12];
  // d_in[13] = previous_visit (==10, fixed by the reference; hardcoded as PV)

  char* ws = (char*)d_ws;
  size_t off = 0;
  auto carve = [&](size_t bytes) -> char* {
    char* p = ws + off;
    off += (bytes + 255) & ~(size_t)255;
    return p;
  };
  bf16_t* Xbf   = (bf16_t*)carve((size_t)Tv * Bv * Fv * 2); // (T,B,F) bf16
  bf16_t* WencT = (bf16_t*)carve((size_t)1024 * 512 * 2);   // [enc_W;enc_U]^T
  bf16_t* WdecT = (bf16_t*)carve((size_t)1024 * 768 * 2);   // [dec_W;dec_U]^T
  bf16_t* d1t   = (bf16_t*)carve((size_t)256 * 256 * 2);
  bf16_t* d2t   = (bf16_t*)carve((size_t)256 * 256 * 2);
  bf16_t* d3t   = (bf16_t*)carve((size_t)256 * 256 * 2);
  bf16_t* hseq  = (bf16_t*)carve((size_t)Tv * BH * 2);      // encoder h per step (bf16)
  bf16_t* h0    = (bf16_t*)carve((size_t)BH * 2);           // zero h
  bf16_t* hd0   = (bf16_t*)carve((size_t)BH * 2);           // decoder h ping
  bf16_t* hd1   = (bf16_t*)carve((size_t)BH * 2);           // decoder h pong
  float*  cbuf  = (float*)carve((size_t)BH * 4);            // cell state (fp32)
  bf16_t* y0    = (bf16_t*)carve((size_t)Bv * Fv * 2);
  bf16_t* y1    = (bf16_t*)carve((size_t)Bv * Fv * 2);

  hipMemsetAsync(h0,   0, (size_t)BH * 2, stream);
  hipMemsetAsync(hd0,  0, (size_t)BH * 2, stream);
  hipMemsetAsync(cbuf, 0, (size_t)BH * 4, stream);

  { int n = Bv * Tv * Fv;
    convert_X_kernel<<<(n + 255) / 256, 256, 0, stream>>>(input, Xbf); }
  transpose_cvt_kernel<<<(256 * 1024 + 255) / 256, 256, 0, stream>>>(enc_W, WencT, 256, 1024, 512, 0);
  transpose_cvt_kernel<<<(256 * 1024 + 255) / 256, 256, 0, stream>>>(enc_U, WencT, 256, 1024, 512, 256);
  transpose_cvt_kernel<<<(512 * 1024 + 255) / 256, 256, 0, stream>>>(dec_W, WdecT, 512, 1024, 768, 0);
  transpose_cvt_kernel<<<(256 * 1024 + 255) / 256, 256, 0, stream>>>(dec_U, WdecT, 256, 1024, 768, 512);
  transpose_cvt_kernel<<<(256 * 256 + 255) / 256, 256, 0, stream>>>(d1_W, d1t, 256, 256, 256, 0);
  transpose_cvt_kernel<<<(256 * 256 + 255) / 256, 256, 0, stream>>>(d2_W, d2t, 256, 256, 256, 0);
  transpose_cvt_kernel<<<(256 * 256 + 255) / 256, 256, 0, stream>>>(d3_W, d3t, 256, 256, 256, 0);

  dim3 grid(8, 16);

  // encoder: 50 steps, K=512 ([x|h]); h written to hseq[t] (double-buffer via slots)
  for (int t = 0; t < Tv; ++t) {
    const bf16_t* hprev = (t == 0) ? h0 : (hseq + (size_t)(t - 1) * BH);
    lstm_step_kernel<<<grid, 256, 0, stream>>>(
        Xbf + (size_t)t * BH, hprev, nullptr,
        WencT, 512, enc_b, cbuf, hseq + (size_t)t * BH);
  }

  // decoder: 41 steps, K=768 ([x|h_enc|h]); fresh zero carry
  hipMemsetAsync(cbuf, 0, (size_t)BH * 4, stream);
  const int nsteps = Tv - (PV - 1); // 41
  for (int s = 0; s < nsteps; ++s) {
    const int t = (PV - 1) + s;
    const bf16_t* hprev = (s & 1) ? hd1 : hd0;
    bf16_t* hnext = (s & 1) ? hd0 : hd1;
    lstm_step_kernel<<<grid, 256, 0, stream>>>(
        Xbf + (size_t)t * BH, hseq + (size_t)t * BH, hprev,
        WdecT, 768, dec_b, cbuf, hnext);
  }
  const bf16_t* hlast = (nsteps & 1) ? hd1 : hd0;

  // dense tower only for the final step (only ys[-1] is returned)
  dense_relu_kernel<<<grid, 256, 0, stream>>>(hlast, d1t, d1_b, y0, nullptr);
  dense_relu_kernel<<<grid, 256, 0, stream>>>(y0,    d2t, d2_b, y1, nullptr);
  dense_relu_kernel<<<grid, 256, 0, stream>>>(y1,    d3t, d3_b, nullptr, (float*)d_out);
}
